// ScannedLSTM_70574902608125
// MI455X (gfx1250) — matrix operand order repaired and checked by reading, not compiled, not run on hardware
//
#include <hip/hip_runtime.h>
#include <math.h>

// ---------------------------------------------------------------------------
// Scanned LSTM for MI455X (gfx1250, wave32, WMMA + async LDS staging).
//   T=1024, B=256, H=512, gates = [x_t | h] @ [Wx ; Wh] + b   (K = 1024)
// Per step: GEMM [256,1024]x[1024,2048] via v_wmma_f32_16x16x32_f16.
// Workgroup = 16 waves = all 16 M-tiles sharing one N-group, so the
// B-fragments are staged through LDS once per block (16x less L2 traffic)
// using global_load_async_to_lds_b64 (ASYNCcnt) with double buffering.
// LSTM pointwise epilogue fused in-register (gate-aligned tiling).
// ---------------------------------------------------------------------------

typedef __attribute__((ext_vector_type(16))) _Float16 v16h;
typedef __attribute__((ext_vector_type(8)))  _Float16 v8h;
typedef __attribute__((ext_vector_type(8)))  float    v8f;
typedef __attribute__((ext_vector_type(4)))  float    f4v;
typedef __attribute__((ext_vector_type(2)))  int      i2v;
typedef __attribute__((ext_vector_type(2)))  unsigned int u2v;

#define TSTEPS 1024
#define NB     256     // batch
#define NH     512     // hidden
#define NG     2048    // 4*H
#define NKK    32      // K blocks of 32 (fused K = 1024)
#define NTILES 128     // N tiles of 16 (2048/16)
#define LSTRIDE 24     // halves per lane fragment in LDS (16 + 8 pad, 48B: keeps
                       // 16B alignment and makes ds_load_b128 bank-conflict-free)

#if defined(__gfx1250__) && __has_builtin(__builtin_amdgcn_global_load_async_to_lds_b64)
#define HAVE_ASYNC_LDS 1
#else
#define HAVE_ASYNC_LDS 0
#endif

#if __has_builtin(__builtin_amdgcn_s_wait_asynccnt)
#define WAIT_ASYNC0() __builtin_amdgcn_s_wait_asynccnt(0)
#else
#define WAIT_ASYNC0() asm volatile("s_wait_asynccnt 0x0" ::: "memory")
#endif

__device__ __forceinline__ float sigm(float v) { return 1.0f / (1.0f + __expf(-v)); }

// ---------------------------------------------------------------------------
// Pack Wx (kk 0..15) and Wh (kk 16..31) into the WMMA B-fragment lane layout:
//   wpack[kk][ntile][lane][v] = W[kk*32 + (lane>=16)*8 + v + (v&8)][ntile*16 + (lane&15)]
// ---------------------------------------------------------------------------
__global__ __launch_bounds__(256)
void pack_weights(const float* __restrict__ Wx, const float* __restrict__ Wh,
                  _Float16* __restrict__ wpack) {
    int tid   = blockIdx.x * 256 + threadIdx.x;   // 0 .. 131071
    int lane  = tid & 31;
    int ntile = (tid >> 5) & 127;
    int kk    = tid >> 12;                        // 0..31
    int col   = ntile * 16 + (lane & 15);
    int khi   = (lane >> 4) << 3;                 // 0 or 8
    const float* Wsrc = (kk < 16) ? Wx : Wh;
    int krow0 = (kk & 15) * 32 + khi;
    _Float16* dst = wpack + ((((size_t)kk * NTILES + ntile) * 32 + lane) << 4);
#pragma unroll
    for (int v = 0; v < 16; ++v)
        dst[v] = (_Float16)Wsrc[(size_t)(krow0 + v + (v & 8)) * NG + col];
}

__global__ __launch_bounds__(256)
void init_state(_Float16* __restrict__ h16a, _Float16* __restrict__ h16b,
                float* __restrict__ cst) {
    int i = blockIdx.x * 256 + threadIdx.x;       // 131072 = B*H
    h16a[i] = (_Float16)0.0f;
    h16b[i] = (_Float16)0.0f;
    cst[i]  = 0.0f;
}

// Stage the 4 gate B-fragments (4 KB) of K-block kk into LDS buffer `dst`.
// 512 threads each move 8 bytes. Async (ASYNCcnt) when available.
__device__ __forceinline__ void stage_bfrags(const _Float16* __restrict__ wpack,
                                             _Float16* __restrict__ dst,
                                             int kk, int nidx, int tid) {
    const int g    = tid >> 7;        // gate 0..3
    const int ln   = (tid >> 2) & 31; // fragment lane
    const int part = tid & 3;         // 8B chunk of the 32B fragment
    const _Float16* src =
        wpack + ((((size_t)kk * NTILES + (g * 32 + nidx)) * 32 + ln) << 4) + part * 4;
    _Float16* d = dst + (g * 32 + ln) * LSTRIDE + part * 4;
#if HAVE_ASYNC_LDS
    __builtin_amdgcn_global_load_async_to_lds_b64(
        (__attribute__((address_space(1))) i2v*)(void*)src,
        (__attribute__((address_space(3))) i2v*)(void*)d, 0, 0);
#else
    *(u2v*)d = *(const u2v*)src;   // VGPR staging fallback (ds_store_b64)
#endif
}

// ---------------------------------------------------------------------------
// One LSTM time step. Grid: 32 blocks x 512 threads (16 waves).
// Wave wv handles M-tile m = wv; whole block shares N-group nidx = blockIdx.x,
// computing the SAME 16 columns of all four gates (N-tiles nidx, +32, +64, +96)
// so the epilogue is in-register and B-fragments are shared via LDS.
// ---------------------------------------------------------------------------
__global__ __launch_bounds__(512)
void lstm_step(int t,
               const float* __restrict__ x,              // [T,B,H] fp32
               const unsigned char* __restrict__ resets, // [T,B] (jnp.bool_, 1B)
               const _Float16* __restrict__ wpack,       // packed [Wx;Wh] f16
               const float* __restrict__ bias,           // [4H]
               const _Float16* __restrict__ h_in,        // [B,H] f16 (prev h)
               _Float16* __restrict__ h_out,             // [B,H] f16 (next h)
               float* __restrict__ cstate,               // [B,H] fp32 cell
               float* __restrict__ ys) {                 // [T,B,H] fp32 output
    __shared__ _Float16 lbuf[2][4 * 32 * LSTRIDE];       // 2 x 6 KB double buffer

    const int tid  = threadIdx.x;
    const int lane = tid & 31;
    const int wv   = tid >> 5;              // 0..15 = M-tile
    const int m    = wv;
    const int nidx = blockIdx.x;            // 0..31 = per-gate column tile
    const int hi   = lane >> 4;             // half-wave select
    const int arow = m * 16 + (lane & 15);  // A-matrix row loaded by this lane

    // Reset mask for the h-part of the A matrix (multiply, keep EXEC all-1s).
    const _Float16 hmask =
        resets[(size_t)t * NB + arow] ? (_Float16)0.0f : (_Float16)1.0f;

    const float*    xrow = x + ((size_t)t * NB + arow) * NH;
    const _Float16* hrow = h_in + (size_t)arow * NH;

    v8f acc_i = {}, acc_f = {}, acc_g = {}, acc_o = {};

    // Prologue: stage K-block 0
    stage_bfrags(wpack, lbuf[0], 0, nidx, tid);
#if HAVE_ASYNC_LDS
    WAIT_ASYNC0();
#endif
    __syncthreads();

#pragma unroll
    for (int kk = 0; kk < NKK; ++kk) {
        const int cur = kk & 1;
        // Prefetch next K-block's B-fragments into the other buffer
        if (kk + 1 < NKK)
            stage_bfrags(wpack, lbuf[1 - cur], kk + 1, nidx, tid);

        // ---- A fragment (16x32 f16): K = kb..kb+7 and kb+16..kb+23,
        //      kb = kk*32 + hi*8 (ISA A layout).
        v16h a;
        if (kk < 16) {
            // x part: fp32 -> f16 convert on the fly
            const int kb = kk * 32 + hi * 8;
            f4v f0 = *(const f4v*)(xrow + kb);
            f4v f1 = *(const f4v*)(xrow + kb + 4);
            f4v f2 = *(const f4v*)(xrow + kb + 16);
            f4v f3 = *(const f4v*)(xrow + kb + 20);
#pragma unroll
            for (int i = 0; i < 4; ++i) {
                a[i]      = (_Float16)f0[i];
                a[4 + i]  = (_Float16)f1[i];
                a[8 + i]  = (_Float16)f2[i];
                a[12 + i] = (_Float16)f3[i];
            }
        } else {
            // h part: contiguous f16 loads + reset mask
            const int kb = (kk - 16) * 32 + hi * 8;
            v8h h0 = *(const v8h*)(hrow + kb);
            v8h h1 = *(const v8h*)(hrow + kb + 16);
            a = __builtin_shufflevector(h0, h1, 0, 1, 2, 3, 4, 5, 6, 7,
                                        8, 9, 10, 11, 12, 13, 14, 15);
            a = a * hmask;
        }

        // ---- 4 B fragments from LDS (bank-conflict-free, 16B aligned)
        const _Float16* lb = &lbuf[cur][0];
#pragma unroll
        for (int g = 0; g < 4; ++g) {
            const _Float16* p = lb + (g * 32 + lane) * LSTRIDE;
            v8h b0 = *(const v8h*)(p);
            v8h b1 = *(const v8h*)(p + 8);
            v16h bf = __builtin_shufflevector(b0, b1, 0, 1, 2, 3, 4, 5, 6, 7,
                                              8, 9, 10, 11, 12, 13, 14, 15);
            if (g == 0)
                acc_i = __builtin_amdgcn_wmma_f32_16x16x32_f16(false, a, false, bf,
                                                               (short)0, acc_i, false, false);
            else if (g == 1)
                acc_f = __builtin_amdgcn_wmma_f32_16x16x32_f16(false, a, false, bf,
                                                               (short)0, acc_f, false, false);
            else if (g == 2)
                acc_g = __builtin_amdgcn_wmma_f32_16x16x32_f16(false, a, false, bf,
                                                               (short)0, acc_g, false, false);
            else
                acc_o = __builtin_amdgcn_wmma_f32_16x16x32_f16(false, a, false, bf,
                                                               (short)0, acc_o, false, false);
        }

        // Finish the prefetch + make it visible; also fences readers before
        // the buffer gets overwritten by the next iteration's prefetch.
        if (kk + 1 < NKK) {
#if HAVE_ASYNC_LDS
            WAIT_ASYNC0();
#endif
            __syncthreads();
        }
    }

    // ---- Fused LSTM pointwise epilogue, fully in-register.
    // C/D layout: VGPR r, lanes 0-15 -> M=r, lanes 16-31 -> M=r+8; N = lane&15.
    const int col = nidx * 16 + (lane & 15);
    const float bias_i = bias[0 * NH + col];
    const float bias_f = bias[1 * NH + col];
    const float bias_g = bias[2 * NH + col];
    const float bias_o = bias[3 * NH + col];

#pragma unroll
    for (int r = 0; r < 8; ++r) {
        const int    brow = m * 16 + hi * 8 + r;
        const size_t idx  = (size_t)brow * NH + col;
        const bool   rst  = resets[(size_t)t * NB + brow] != 0;

        float iv = sigm(acc_i[r] + bias_i);
        float fv = sigm(acc_f[r] + bias_f);
        float gv = tanhf(acc_g[r] + bias_g);
        float ov = sigm(acc_o[r] + bias_o);

        float cprev = rst ? 0.0f : cstate[idx];
        float cnew  = fv * cprev + iv * gv;
        float hnew  = ov * tanhf(cnew);

        cstate[idx] = cnew;
        ys[(size_t)t * NB * NH + idx] = hnew;
        h_out[idx] = (_Float16)hnew;
    }
}

// ---------------------------------------------------------------------------
// Host launch: pack weights, zero state, then 1024 sequential step kernels
// (stream order provides the scan dependency; graph replay amortizes launch).
// Workspace: wpack 4MB | h16 ping 256KB | h16 pong 256KB | c 512KB  (~5MB).
// ---------------------------------------------------------------------------
extern "C" void kernel_launch(void* const* d_in, const int* in_sizes, int n_in,
                              void* d_out, int out_size, void* d_ws, size_t ws_size,
                              hipStream_t stream) {
    const float*         x      = (const float*)d_in[0];
    const unsigned char* resets = (const unsigned char*)d_in[1]; // jnp.bool_ = 1B/elem
    const float*         Wx     = (const float*)d_in[2];
    const float*         Wh     = (const float*)d_in[3];
    const float*         bias   = (const float*)d_in[4];
    float*               ys     = (float*)d_out;

    char* ws = (char*)d_ws;
    _Float16* wpack = (_Float16*)(ws);                                  // 4 MB
    _Float16* h16a  = (_Float16*)(ws + (4u << 20));                     // 256 KB
    _Float16* h16b  = (_Float16*)(ws + (4u << 20) + (256u << 10));      // 256 KB
    float*    cst   = (float*)   (ws + (4u << 20) + (512u << 10));      // 512 KB

    pack_weights<<<512, 256, 0, stream>>>(Wx, Wh, wpack);
    init_state<<<512, 256, 0, stream>>>(h16a, h16b, cst);

    for (int t = 0; t < TSTEPS; ++t) {
        const _Float16* hin  = (t & 1) ? h16b : h16a;
        _Float16*       hout = (t & 1) ? h16a : h16b;
        lstm_step<<<32, 512, 0, stream>>>(t, x, resets, wpack, bias,
                                          hin, hout, cst, ys);
    }
}
